// LSTM_18159121728076
// MI455X (gfx1250) — matrix operand order repaired and checked by reading, not compiled, not run on hardware
//
#include <hip/hip_runtime.h>

typedef __bf16 bf16;
typedef __bf16 v8bf  __attribute__((ext_vector_type(8)));
typedef __bf16 v16bf __attribute__((ext_vector_type(16)));
typedef float  v4f   __attribute__((ext_vector_type(4)));
typedef float  v8f   __attribute__((ext_vector_type(8)));

constexpr int kH  = 64;    // hidden size
constexpr int kT  = 2048;  // sequence length
constexpr int kB  = 256;   // batch
constexpr int k3H = 192;   // 3*H (r,z,n gates)

// fast sigmoid/tanh: one v_exp_f32 + one v_rcp_f32 each.
// Overflow is benign: e=inf -> rcp(inf)=0 -> sigmoid=0 / tanh=-1.
__device__ __forceinline__ float sigmoid_f(float x) {
  return __builtin_amdgcn_rcpf(1.0f + __expf(-x));
}
__device__ __forceinline__ float tanh_f(float x) {
  return 2.0f * __builtin_amdgcn_rcpf(1.0f + __expf(-2.0f * x)) - 1.0f;
}

// CDNA5 async global->LDS DMA (ASYNCcnt-tracked), 16B per lane.
__device__ __forceinline__ void async_load_b128(uint32_t lds_off, const void* gptr) {
  asm volatile("global_load_async_to_lds_b128 %0, %1, off"
               :: "v"(lds_off), "v"((unsigned long long)(uintptr_t)gptr)
               : "memory");
}
__device__ __forceinline__ void wait_async0() {
  asm volatile("s_wait_asynccnt 0x0" ::: "memory");
}

// x[B,H,T] fp32  ->  y[B,T,H] bf16   (coalesced load + coalesced store via LDS tile)
__global__ __launch_bounds__(256)
void gru_transpose_in(const float* __restrict__ x, bf16* __restrict__ y) {
  __shared__ bf16 tile[64][66];  // 132B row stride -> conflict-free
  const int b  = blockIdx.y;
  const int t0 = blockIdx.x * 64;
  for (int i = threadIdx.x; i < 64 * 64; i += 256) {
    int h = i >> 6, tt = i & 63;
    tile[h][tt] = (bf16)x[((size_t)b * kH + h) * kT + t0 + tt];
  }
  __syncthreads();
  for (int i = threadIdx.x; i < 64 * 64; i += 256) {
    int tt = i >> 6, h = i & 63;
    y[((size_t)b * kT + t0 + tt) * kH + h] = tile[h][tt];
  }
}

// z[B,T,H] fp32  ->  out[B,H,T] fp32
__global__ __launch_bounds__(256)
void gru_transpose_out(const float* __restrict__ z, float* __restrict__ out) {
  __shared__ float tile[64][65];
  const int b  = blockIdx.y;
  const int t0 = blockIdx.x * 64;
  for (int i = threadIdx.x; i < 64 * 64; i += 256) {
    int tt = i >> 6, h = i & 63;
    tile[tt][h] = z[((size_t)b * kT + t0 + tt) * kH + h];
  }
  __syncthreads();
  for (int i = threadIdx.x; i < 64 * 64; i += 256) {
    int h = i >> 6, tt = i & 63;
    out[((size_t)b * kH + h) * kT + t0 + tt] = tile[tt][h];
  }
}

// One GRU layer. 128-thread (4-wave) workgroup owns 16 batch rows for the whole
// sequence. Wave w owns hidden columns j in [16w,16w+16): it computes the r/z/n
// gate tiles for exactly those columns, so gate math is purely per-lane.
// r,z gates: x@Wih^T and h@Whh^T chain into ONE accumulator (matrix core adds).
// n gate keeps gx_n / gh_n separate (n = tanh(gx_n + r*gh_n)).
// x_t tiles arrive via double-buffered global_load_async_to_lds_b128 issued one
// step ahead; outputs stream one step delayed from the LDS h mirror (coalesced).
__global__ __launch_bounds__(128)
void gru_layer_wmma(const bf16*  __restrict__ in,    // [B,T,H] bf16
                    const float* __restrict__ Wih,   // [192,64] fp32 (this layer)
                    const float* __restrict__ Whh,   // [192,64]
                    const float* __restrict__ bih,   // [192]
                    const float* __restrict__ bhh,   // [192]
                    bf16*  __restrict__ out_bf,      // [B,T,H] bf16 or null
                    float* __restrict__ out_f)       // [B,T,H] fp32 or null
{
  __shared__ __align__(16) bf16 xbuf[2][16][64];  // double-buffered x tile
  __shared__ __align__(16) bf16 hbuf[16][64];

  const int tid   = threadIdx.x;
  const int wave  = tid >> 5;
  const int lane  = tid & 31;
  const int b0    = blockIdx.x * 16;
  const int N     = lane & 15;   // WMMA N index / A-matrix row index
  const int khalf = lane >> 4;   // which K half this lane holds
  const int j     = wave * 16 + N;  // hidden column owned by this lane

  // ---- preload weight B-fragments as bf16 (resident for all 2048 steps) ----
  // B fragment k order used here (32x16, 16-bit): lane l -> column N=l&15; lanes 0-15 hold K=0..15,
  // lanes 16-31 hold K=16..31, contiguous per lane.
  v16bf wih[3][2], whh[3][2];
  float bi[3], bh[3];
#pragma unroll
  for (int g = 0; g < 3; ++g) {
    const int row = g * kH + j;                 // weight row in [0,192)
    bi[g] = bih[row];
    bh[g] = bhh[row];
    const float* wi = Wih + (size_t)row * kH;
    const float* wh = Whh + (size_t)row * kH;
#pragma unroll
    for (int kb = 0; kb < 2; ++kb) {
      const int k0 = kb * 32 + khalf * 16;
      v16bf fi, fh;
#pragma unroll
      for (int e = 0; e < 16; ++e) {
        fi[e] = (bf16)wi[k0 + e];
        fh[e] = (bf16)wh[k0 + e];
      }
      wih[g][kb] = fi;
      whh[g][kb] = fh;
    }
  }

  // accumulator init vectors (bias broadcasts)
  v8f arz0_i, arz1_i, axn_i, ahn_i;
#pragma unroll
  for (int e = 0; e < 8; ++e) {
    arz0_i[e] = bi[0] + bh[0];
    arz1_i[e] = bi[1] + bh[1];
    axn_i[e]  = bi[2];
    ahn_i[e]  = bh[2];
  }

  // Pin loop-invariant values into VGPRs as opaque asm results so LLVM cannot
  // sink/rematerialize the f32->bf16 packing or bias broadcasts into the loop.
#pragma unroll
  for (int g = 0; g < 3; ++g) {
#pragma unroll
    for (int kb = 0; kb < 2; ++kb) {
      asm volatile("" : "+v"(wih[g][kb]));
      asm volatile("" : "+v"(whh[g][kb]));
    }
  }
  asm volatile("" : "+v"(arz0_i), "+v"(arz1_i), "+v"(axn_i), "+v"(ahn_i));

  const int xrow = tid >> 3;          // 16 rows x 8 chunks of 16B
  const int xchk = tid & 7;
  const bf16* xsrc   = in + (size_t)(b0 + xrow) * kT * kH + xchk * 8;
  const size_t obase = (size_t)(b0 + xrow) * kT * kH + xchk * 8;
  // LDS flat pointers: low 32 bits are the wave-relative LDS byte offset
  const uint32_t la[2] = {
      (uint32_t)(uintptr_t)&xbuf[0][xrow][xchk * 8],
      (uint32_t)(uintptr_t)&xbuf[1][xrow][xchk * 8]};

  // kick off the DMA for t=0 as early as possible
  async_load_b128(la[0], xsrc);

  // ---- h0 = 0 (registers + LDS mirror) ----
  v8f hreg = {};
  for (int i = tid; i < (16 * 64 * 2) / 16; i += 128)
    ((uint4*)hbuf)[i] = uint4{0u, 0u, 0u, 0u};

  for (int t = 0; t < kT; ++t) {
    const int cur = t & 1;
    wait_async0();    // own async writes (x_t, issued last iter) have landed
    __syncthreads();  // (a) -> everyone's x_t writes + prev h writes visible

    // issue DMA for x_{t+1} into the other buffer (overlaps with compute)
    if (t + 1 < kT) {
      async_load_b128(la[cur ^ 1], xsrc + (size_t)(t + 1) * kH);
      if (t + 2 < kT) __builtin_prefetch(xsrc + (size_t)(t + 2) * kH, 0, 1);
    }

    // ---- delayed output store: h_{t-1} sits row-major in hbuf ----
    if (t > 0) {
      v8bf hv = *(const v8bf*)&hbuf[xrow][xchk * 8];
      const size_t o = obase + (size_t)(t - 1) * kH;
      if (out_bf) {
        *(v8bf*)(out_bf + o) = hv;
      } else {
        v4f f0, f1;
#pragma unroll
        for (int e = 0; e < 4; ++e) { f0[e] = (float)hv[e]; f1[e] = (float)hv[4 + e]; }
        *(v4f*)(out_f + o)     = f0;
        *(v4f*)(out_f + o + 4) = f1;
      }
    }

    // A-fragments (16x32 bf16 layout): lane row M=N; e0..15 = K kb*32+16*khalf+0..15
    // (contiguous, matching B).
    const int koff = khalf * 16;
    v16bf xa[2], ha[2];
#pragma unroll
    for (int kb = 0; kb < 2; ++kb) {
      v8bf xlo = *(const v8bf*)&xbuf[cur][N][kb * 32 + koff];
      v8bf xhi = *(const v8bf*)&xbuf[cur][N][kb * 32 + koff + 8];
      v8bf hlo = *(const v8bf*)&hbuf[N][kb * 32 + koff];
      v8bf hhi = *(const v8bf*)&hbuf[N][kb * 32 + koff + 8];
      xa[kb] = __builtin_shufflevector(xlo, xhi, 0,1,2,3,4,5,6,7,8,9,10,11,12,13,14,15);
      ha[kb] = __builtin_shufflevector(hlo, hhi, 0,1,2,3,4,5,6,7,8,9,10,11,12,13,14,15);
    }
    __syncthreads();  // (b) all waves done reading hbuf before rewrite

    // accumulators start from pinned bias vectors; first WMMA of each chain
    // reads them as C (D != C), so no per-step broadcast movs are needed.
    v8f arz[2] = {arz0_i, arz1_i};
    v8f axn = axn_i, ahn = ahn_i;

    // 12 WMMAs total (K=64 = 2x32, two operand matrices)
#pragma unroll
    for (int g = 0; g < 2; ++g) {
#pragma unroll
      for (int kb = 0; kb < 2; ++kb) {
        arz[g] = __builtin_amdgcn_wmma_f32_16x16x32_bf16(
            false, xa[kb], false, wih[g][kb], (short)0, arz[g], false, false);
        arz[g] = __builtin_amdgcn_wmma_f32_16x16x32_bf16(
            false, ha[kb], false, whh[g][kb], (short)0, arz[g], false, false);
      }
    }
#pragma unroll
    for (int kb = 0; kb < 2; ++kb) {
      axn = __builtin_amdgcn_wmma_f32_16x16x32_bf16(
          false, xa[kb], false, wih[2][kb], (short)0, axn, false, false);
      ahn = __builtin_amdgcn_wmma_f32_16x16x32_bf16(
          false, ha[kb], false, whh[2][kb], (short)0, ahn, false, false);
    }

    // gate math, per-lane: element e <-> batch row m = e + 8*khalf, column j
    v8f hnew;
#pragma unroll
    for (int e = 0; e < 8; ++e) {
      float r = sigmoid_f(arz[0][e]);
      float z = sigmoid_f(arz[1][e]);
      float n = tanh_f(axn[e] + r * ahn[e]);
      hnew[e] = (1.0f - z) * n + z * hreg[e];
    }
    hreg = hnew;

    // publish h_t to LDS mirror (consumed next step for A-frag + output store)
#pragma unroll
    for (int e = 0; e < 8; ++e) {
      hbuf[e + 8 * khalf][j] = (bf16)hnew[e];
    }
    // no barrier here: barrier (a) of next iteration orders hbuf writes->reads
  }

  // ---- tail: store h_{T-1} ----
  __syncthreads();
  {
    v8bf hv = *(const v8bf*)&hbuf[xrow][xchk * 8];
    const size_t o = obase + (size_t)(kT - 1) * kH;
    if (out_bf) {
      *(v8bf*)(out_bf + o) = hv;
    } else {
      v4f f0, f1;
#pragma unroll
      for (int e = 0; e < 4; ++e) { f0[e] = (float)hv[e]; f1[e] = (float)hv[4 + e]; }
      *(v4f*)(out_f + o)     = f0;
      *(v4f*)(out_f + o + 4) = f1;
    }
  }
}

extern "C" void kernel_launch(void* const* d_in, const int* in_sizes, int n_in,
                              void* d_out, int out_size, void* d_ws, size_t ws_size,
                              hipStream_t stream) {
  (void)in_sizes; (void)n_in; (void)out_size; (void)ws_size;
  const float* x   = (const float*)d_in[0];   // [B, H, T]
  const float* Wih = (const float*)d_in[1];   // [L, 3H, H]
  const float* Whh = (const float*)d_in[2];   // [L, 3H, H]
  const float* bih = (const float*)d_in[3];   // [L, 3H]
  const float* bhh = (const float*)d_in[4];   // [L, 3H]
  float* out = (float*)d_out;                 // [B, H, T]

  // workspace layout (needs 256 MiB):
  //   buf0 bf16 [B,T,H]  @ 0       (64 MiB)
  //   buf1 bf16 [B,T,H]  @ 64 MiB  (64 MiB)
  //   buf2 f32  [B,T,H]  @ 128 MiB (128 MiB)
  const size_t BTH = (size_t)kB * kT * kH;
  char* ws = (char*)d_ws;
  bf16*  buf0 = (bf16*)ws;
  bf16*  buf1 = (bf16*)(ws + BTH * 2);
  float* buf2 = (float*)(ws + BTH * 4);

  dim3 tgrid(kT / 64, kB);
  gru_transpose_in<<<tgrid, 256, 0, stream>>>(x, buf0);

  const size_t WL = (size_t)k3H * kH;  // per-layer weight matrix elements
  gru_layer_wmma<<<kB / 16, 128, 0, stream>>>(
      buf0, Wih + 0 * WL, Whh + 0 * WL, bih + 0 * k3H, bhh + 0 * k3H, buf1, nullptr);
  gru_layer_wmma<<<kB / 16, 128, 0, stream>>>(
      buf1, Wih + 1 * WL, Whh + 1 * WL, bih + 1 * k3H, bhh + 1 * k3H, buf0, nullptr);
  gru_layer_wmma<<<kB / 16, 128, 0, stream>>>(
      buf0, Wih + 2 * WL, Whh + 2 * WL, bih + 2 * k3H, bhh + 2 * k3H, nullptr, buf2);

  gru_transpose_out<<<tgrid, 256, 0, stream>>>(buf2, out);
}
